// GatedLSTM_10187662426297
// MI455X (gfx1250) — matrix operand order repaired and checked by reading, not compiled, not run on hardware
//
#include <hip/hip_runtime.h>
#include <hip/hip_bf16.h>

typedef __attribute__((ext_vector_type(16))) _Float16 v16h;
typedef __attribute__((ext_vector_type(8)))  _Float16 v8h;
typedef __attribute__((ext_vector_type(8)))  float    v8f;

constexpr int kB = 32;    // batch
constexpr int kS = 512;   // sequence
constexpr int kI = 512;   // input dim
constexpr int kH = 1024;  // hidden dim

static __device__ __forceinline__ v8f wmma_f16(v16h a, v16h b, v8f c) {
  // D = A(16x32 f16) * B(32x16 f16) + C(16x16 f32)
  return __builtin_amdgcn_wmma_f32_16x16x32_f16(
      /*neg_a=*/false, a, /*neg_b=*/false, b,
      /*c_mod=*/(short)0, c, /*reuse_a=*/false, /*reuse_b=*/false);
}

static __device__ __forceinline__ v16h load_a_tile(const _Float16* row, int kc, int hh) {
  v16h a;
  ((v8h*)&a)[0] = *(const v8h*)(row + kc + hh * 16);
  ((v8h*)&a)[1] = *(const v8h*)(row + kc + hh * 16 + 8);
  return a;
}

// ---------------------------------------------------------------------------
// f32 -> f16 conversion (grid-stride)
// ---------------------------------------------------------------------------
__global__ void k_cvt_f16(const float* __restrict__ src, _Float16* __restrict__ dst,
                          long n) {
  long i = (long)blockIdx.x * blockDim.x + threadIdx.x;
  long stride = (long)gridDim.x * blockDim.x;
  for (; i < n; i += stride) dst[i] = (_Float16)src[i];
}

// ---------------------------------------------------------------------------
// Transpose K x N f32 (row-major) -> N x K f16 (row-major). One-time cost;
// makes every WMMA B-tile a contiguous 32B load per lane.
// ---------------------------------------------------------------------------
__global__ void k_transpose_f16(const float* __restrict__ W, _Float16* __restrict__ WT,
                                int K, int N) {
  long total = (long)K * N;
  long i = (long)blockIdx.x * blockDim.x + threadIdx.x;
  long stride = (long)gridDim.x * blockDim.x;
  for (; i < total; i += stride) {
    long k = i / N, n = i % N;
    WT[n * (long)K + k] = (_Float16)W[i];
  }
}

// ---------------------------------------------------------------------------
// Input projections: xpre[g][t][b][n] = sum_k x[b][t][k] * Wx[g][k][n] + bias[g][n]
// Grid: (16 n-groups of 4 tiles, 1024 m-tiles, 4 gates). One wave per block;
// 4 independent accumulator chains; software double-buffered operand loads.
// launch_bounds(32,1): give the RA a full register budget so both operand
// generations stay live (no loadcnt-0 drain per WMMA).
// ---------------------------------------------------------------------------
__global__ void __launch_bounds__(32, 1)
k_xproj(const _Float16* __restrict__ xh,   // [B*S, I] f16
        const _Float16* __restrict__ WxT,  // [4][H][I] f16 (N-major)
        const float* __restrict__ bias,    // [4][H] f32
        float* __restrict__ xpre) {        // [4][S][B][H] f32
  const int gate  = blockIdx.z;
  const int mtile = blockIdx.y;
  const int n0    = blockIdx.x * 4;
  const int lane  = threadIdx.x & 31;
  const int hh    = lane >> 4;      // lane half (K-split of the WMMA layout)
  const int mr    = lane & 15;      // A row within tile / B column within tile

  const int r = mtile * 16 + mr;    // global x row = b*S + t
  const _Float16* arow = xh + (size_t)r * kI;
  const _Float16* brow = WxT + ((size_t)gate * kH + (size_t)n0 * 16 + mr) * kI;

  v8f acc[4] = {};
  v16h a = load_a_tile(arow, 0, hh);
  v16h bm[4];
#pragma unroll
  for (int j = 0; j < 4; ++j)
    bm[j] = *(const v16h*)(brow + (size_t)j * 16 * kI + hh * 16);

#pragma unroll
  for (int kc = 0; kc < kI; kc += 32) {
    v16h a_n;
    v16h bm_n[4];
    if (kc + 32 < kI) {  // issue next chunk's loads before current WMMAs
      a_n = load_a_tile(arow, kc + 32, hh);
#pragma unroll
      for (int j = 0; j < 4; ++j)
        bm_n[j] = *(const v16h*)(brow + (size_t)j * 16 * kI + kc + 32 + hh * 16);
    }
#pragma unroll
    for (int j = 0; j < 4; ++j)
      acc[j] = wmma_f16(a, bm[j], acc[j]);
    if (kc + 32 < kI) {
      a = a_n;
#pragma unroll
      for (int j = 0; j < 4; ++j) bm[j] = bm_n[j];
    }
  }

#pragma unroll
  for (int j = 0; j < 4; ++j) {
    const int n  = (n0 + j) * 16 + mr;            // C layout: N = lane%16
    const float bv = bias[gate * kH + n];
#pragma unroll
    for (int rr = 0; rr < 8; ++rr) {
      const int rowg = mtile * 16 + rr + hh * 8;  // C layout: M = reg + 8*half
      const int b = rowg >> 9;                    // S = 512
      const int t = rowg & (kS - 1);
      xpre[(((size_t)gate * kS + t) * kB + b) * kH + n] = acc[j][rr] + bv;
    }
  }
}

// ---------------------------------------------------------------------------
// One LSTM timestep. Block = 128 threads (4 waves). The K=1024 reduction is
// split across the 4 waves (256 each); every wave computes ALL 4 gates for
// its K-slice (4 independent WMMA accumulator chains, A-tile reused across
// gates). Double-buffered operand loads; launch_bounds(128,1) gives the RA
// headroom to keep both operand generations live (only 512 waves exist per
// step GPU-wide, so occupancy is irrelevant — per-wave pipelining is the
// only latency lever). Partial sums reduce through LDS in the fused
// gate/cell-state update. Grid: (64 n-tiles, 2 m-tiles).
// ---------------------------------------------------------------------------
__global__ void __launch_bounds__(128, 1)
k_step(const _Float16* __restrict__ hprev,  // [B][H] f16
       _Float16* __restrict__ hnext,        // [B][H] f16
       float* __restrict__ c_state,         // [B][H] f32
       const _Float16* __restrict__ WhT,    // [4][H][H] f16 (N-major)
       const float* __restrict__ xpre,      // [4][S][B][H] f32
       float* __restrict__ out, int t) {
  __shared__ float lds[16 * 256];  // [gate][kwave][m*16+n] partial sums (16 KB)

  const int tid  = threadIdx.x;
  const int wv   = tid >> 5;        // K-slice wave id (0..3)
  const int lane = tid & 31;
  const int hh   = lane >> 4;
  const int mr   = lane & 15;
  const int ntile = blockIdx.x;
  const int mtile = blockIdx.y;

  const _Float16* arow = hprev + (size_t)(mtile * 16 + mr) * kH + wv * 256;
  const _Float16* brow = WhT + (size_t)(ntile * 16 + mr) * kH + wv * 256;
  constexpr size_t kGateW = (size_t)kH * kH;  // gate stride in WhT
  constexpr int kKS = kH / 4;                 // K-slice per wave = 256

  v8f acc[4] = {};
  v16h a = load_a_tile(arow, 0, hh);
  v16h bm[4];
#pragma unroll
  for (int g = 0; g < 4; ++g)
    bm[g] = *(const v16h*)(brow + (size_t)g * kGateW + hh * 16);

#pragma unroll
  for (int kc = 0; kc < kKS; kc += 32) {
    v16h a_n;
    v16h bm_n[4];
    if (kc + 32 < kKS) {  // issue next chunk's loads before current WMMAs
      a_n = load_a_tile(arow, kc + 32, hh);
#pragma unroll
      for (int g = 0; g < 4; ++g)
        bm_n[g] = *(const v16h*)(brow + (size_t)g * kGateW + kc + 32 + hh * 16);
    }
#pragma unroll
    for (int g = 0; g < 4; ++g)
      acc[g] = wmma_f16(a, bm[g], acc[g]);
    if (kc + 32 < kKS) {
      a = a_n;
#pragma unroll
      for (int g = 0; g < 4; ++g) bm[g] = bm_n[g];
    }
  }

#pragma unroll
  for (int g = 0; g < 4; ++g)
#pragma unroll
    for (int rr = 0; rr < 8; ++rr)
      lds[(g * 4 + wv) * 256 + (rr + hh * 8) * 16 + mr] = acc[g][rr];
  __syncthreads();

  const size_t SBH = (size_t)kS * kB * kH;
  const size_t BH  = (size_t)kB * kH;
  const size_t BSH = (size_t)kB * kS * kH;
  float* fs = out + BSH + 2 * BH;  // [hidden_seq][hT][cT][fs][is][os][gs][cs]

  for (int e = tid; e < 256; e += 128) {
    const int m = e >> 4, n = e & 15;
    const int b = mtile * 16 + m;
    const int h = ntile * 16 + n;
    const size_t gb = ((size_t)t * kB + b) * kH + h;

    float s[4];
#pragma unroll
    for (int g = 0; g < 4; ++g) {
      const float* p = lds + g * 1024 + e;
      s[g] = (p[0] + p[256]) + (p[512] + p[768]);  // reduce 4 K-slices
    }
    const float vi = s[0] + xpre[gb];
    const float vf = s[1] + xpre[gb + SBH];
    const float vg = s[2] + xpre[gb + 2 * SBH];
    const float vo = s[3] + xpre[gb + 3 * SBH];

    const float it = 1.0f / (1.0f + __expf(-vi));
    const float ft = 1.0f / (1.0f + __expf(-vf));
    const float gt = tanhf(vg);
    const float ot = 1.0f / (1.0f + __expf(-vo));

    const size_t ci = (size_t)b * kH + h;
    const float cp = c_state[ci];
    const float cn = ft * cp + it * gt;
    const float hn = ot * tanhf(cn);

    c_state[ci] = cn;
    hnext[ci] = (_Float16)hn;
    out[((size_t)b * kS + t) * kH + h] = hn;  // hidden_seq (B,S,H)
    fs[gb]           = ft;                    // fs (S,B,H)
    fs[gb + SBH]     = it;                    // is
    fs[gb + 2 * SBH] = ot;                    // os
    fs[gb + 3 * SBH] = gt;                    // gs
    fs[gb + 4 * SBH] = cn;                    // cs
  }
}

// ---------------------------------------------------------------------------
// hT = hidden_seq[:, S-1, :], cT = c_state
// ---------------------------------------------------------------------------
__global__ void k_final(const float* __restrict__ c_state, float* __restrict__ out) {
  const int idx = blockIdx.x * blockDim.x + threadIdx.x;
  if (idx >= kB * kH) return;
  const int b = idx / kH, h = idx % kH;
  const size_t BSH = (size_t)kB * kS * kH;
  out[BSH + idx] = out[((size_t)b * kS + (kS - 1)) * kH + h];
  out[BSH + (size_t)kB * kH + idx] = c_state[idx];
}

extern "C" void kernel_launch(void* const* d_in, const int* in_sizes, int n_in,
                              void* d_out, int out_size, void* d_ws, size_t ws_size,
                              hipStream_t stream) {
  (void)in_sizes; (void)n_in; (void)out_size; (void)ws_size;

  const float* x  = (const float*)d_in[0];
  const float* h0 = (const float*)d_in[1];
  const float* c0 = (const float*)d_in[2];
  const float* Wx[4] = { (const float*)d_in[3],  (const float*)d_in[6],
                         (const float*)d_in[9],  (const float*)d_in[12] };
  const float* Wh[4] = { (const float*)d_in[4],  (const float*)d_in[7],
                         (const float*)d_in[10], (const float*)d_in[13] };
  const float* bb[4] = { (const float*)d_in[5],  (const float*)d_in[8],
                         (const float*)d_in[11], (const float*)d_in[14] };
  float* out = (float*)d_out;

  // Workspace carve-up (~285 MB total)
  char* w = (char*)d_ws;
  _Float16* xh   = (_Float16*)w; w += (size_t)kB * kS * kI * 2;     // 16 MB
  _Float16* WxT  = (_Float16*)w; w += (size_t)4 * kH * kI * 2;      //  4 MB
  _Float16* WhT  = (_Float16*)w; w += (size_t)4 * kH * kH * 2;      //  8 MB
  float*    xpre = (float*)w;    w += (size_t)4 * kS * kB * kH * 4; // 256 MB
  float*    bias = (float*)w;    w += (size_t)4 * kH * 4;
  _Float16* hbuf0 = (_Float16*)w; w += (size_t)kB * kH * 2;
  _Float16* hbuf1 = (_Float16*)w; w += (size_t)kB * kH * 2;
  float*    c_state = (float*)w;  w += (size_t)kB * kH * 4;

  // --- one-time precision / layout conversion ---
  {
    long n = (long)kB * kS * kI;
    k_cvt_f16<<<(int)((n + 255) / 256), 256, 0, stream>>>(x, xh, n);
  }
  for (int g = 0; g < 4; ++g) {
    long n1 = (long)kI * kH;
    k_transpose_f16<<<(int)((n1 + 255) / 256), 256, 0, stream>>>(
        Wx[g], WxT + (size_t)g * kH * kI, kI, kH);
    long n2 = (long)kH * kH;
    k_transpose_f16<<<(int)((n2 + 255) / 256), 256, 0, stream>>>(
        Wh[g], WhT + (size_t)g * kH * kH, kH, kH);
    hipMemcpyAsync(bias + (size_t)g * kH, bb[g], kH * sizeof(float),
                   hipMemcpyDeviceToDevice, stream);
  }
  {
    long n = (long)kB * kH;
    k_cvt_f16<<<(int)((n + 255) / 256), 256, 0, stream>>>(h0, hbuf0, n);
    hipMemcpyAsync(c_state, c0, (size_t)kB * kH * sizeof(float),
                   hipMemcpyDeviceToDevice, stream);
  }

  // --- fused input projections for all 4 gates (big WMMA GEMM) ---
  {
    dim3 grid(16, (kB * kS) / 16, 4);  // (n-groups, m-tiles, gates)
    k_xproj<<<grid, 32, 0, stream>>>(xh, WxT, bias, xpre);
  }

  // --- sequential scan: 512 graph-captured step kernels, ping-pong h ---
  for (int t = 0; t < kS; ++t) {
    const _Float16* hin = (t & 1) ? hbuf1 : hbuf0;
    _Float16* hout      = (t & 1) ? hbuf0 : hbuf1;
    dim3 grid(kH / 16, kB / 16);  // (64 n-tiles, 2 m-tiles)
    k_step<<<grid, 128, 0, stream>>>(hin, hout, c_state, WhT, xpre, out, t);
  }

  // --- hT / cT ---
  k_final<<<(kB * kH + 255) / 256, 256, 0, stream>>>(c_state, out);
}
